// StandardAttention_78915729097245
// MI455X (gfx1250) — compile-verified, matrix operand order checked by reading, not run on hardware
//
#include <hip/hip_runtime.h>
#include <hip/hip_bf16.h>
#include <cstdint>

// ---------------------------------------------------------------------------
// Types for WMMA fragments (CDNA5 / gfx1250, wave32)
// ---------------------------------------------------------------------------
typedef _Float16 half16 __attribute__((ext_vector_type(16))); // A/B operand, 8 VGPRs
typedef _Float16 half8  __attribute__((ext_vector_type(8)));  // 16-byte load unit
typedef float    float8 __attribute__((ext_vector_type(8)));  // C/D accumulator

__device__ __forceinline__ float8 wmma_f16(half16 a, half16 b, float8 c) {
    // D = A(16x32) * B(32x16) + C, f32 accumulate
    return __builtin_amdgcn_wmma_f32_16x16x32_f16(
        /*neg_a=*/false, a, /*neg_b=*/false, b,
        /*c_mod=*/(short)0, c, /*reuse_a=*/false, /*reuse_b=*/false);
}

// 16-bit A-matrix 16x32 layout (ISA 7.12.2): lane l16 = row M, half = lane>>4.
// VGPR0..3 hold K = 8*half + [0..7], VGPR4..7 hold K = 16 + 8*half + [0..7].
// => per lane: two contiguous 8-half (16B) chunks of row M.
__device__ __forceinline__ half16 load_frag_row(const _Float16* base, int stride,
                                                int l16, int hv) {
    const half8 lo = *(const half8*)(base + l16 * stride + 8 * hv);
    const half8 hi = *(const half8*)(base + l16 * stride + 16 + 8 * hv);
    return __builtin_shufflevector(lo, hi, 0,1,2,3,4,5,6,7,8,9,10,11,12,13,14,15);
}

// Cross-lane reductions within 16-lane halves (matches WMMA C-layout rows)
__device__ __forceinline__ float redmax16(float v) {
    v = fmaxf(v, __shfl_xor(v, 1, 16));
    v = fmaxf(v, __shfl_xor(v, 2, 16));
    v = fmaxf(v, __shfl_xor(v, 4, 16));
    v = fmaxf(v, __shfl_xor(v, 8, 16));
    return v;
}
__device__ __forceinline__ float redsum16(float v) {
    v += __shfl_xor(v, 1, 16);
    v += __shfl_xor(v, 2, 16);
    v += __shfl_xor(v, 4, 16);
    v += __shfl_xor(v, 8, 16);
    return v;
}

// ---------------------------------------------------------------------------
// Problem constants
// ---------------------------------------------------------------------------
#define B_      8
#define C_      512
#define HW_     1024     // 32*32 tokens
#define HEADS_  8
#define DH_     64
#define INNER_  512
#define GROUPS_ 32
#define LDS_STRIDE 48            // padded row stride (96B, 16B-aligned)
#define TILE_HALFS (128 * LDS_STRIDE)

// ---------------------------------------------------------------------------
// CDNA5 async global->LDS copy (ASYNCcnt path). One 16B packet per lane.
// ---------------------------------------------------------------------------
__device__ __forceinline__ void async_copy_b128(const _Float16* gp, _Float16* lp) {
    const unsigned long long ga = (unsigned long long)(uintptr_t)gp;
    const unsigned la = (unsigned)(uintptr_t)lp;   // flat->LDS: addr[31:0]
    asm volatile("global_load_async_to_lds_b128 %0, %1, off"
                 :: "v"(la), "v"(ga) : "memory");
}
__device__ __forceinline__ void async_wait0() {
    asm volatile("s_wait_asynccnt 0" ::: "memory");
}

// Stage one 128x32 f16 tile (row-major, K contiguous) into LDS asynchronously.
__device__ __forceinline__ void async_tile_load(const _Float16* __restrict__ g,
                                                int K, int blk, int k0,
                                                _Float16* lds) {
    const int tid = threadIdx.x;
    for (int c = tid; c < 512; c += 256) {          // 128 rows x 4 16B-chunks
        const int row = c >> 2;
        const int col = (c & 3) << 3;
        async_copy_b128(g + (size_t)(blk + row) * K + k0 + col,
                        lds + row * LDS_STRIDE + col);
    }
}

// ---------------------------------------------------------------------------
// Kernel 1: GroupNorm -> f16, token-major [b, p, c]
// ---------------------------------------------------------------------------
__global__ void gn_kernel(const float* __restrict__ x,
                          const float* __restrict__ gamma,
                          const float* __restrict__ beta,
                          _Float16* __restrict__ xh) {
    __shared__ float r1[256];
    __shared__ float r2[256];
    const int tid = threadIdx.x;
    const int bg  = blockIdx.x;
    const int b   = bg >> 5;          // 32 groups
    const int g   = bg & 31;
    const float* xb = x + ((size_t)b * C_ + g * 16) * HW_;   // 16 ch x 1024 px

    float s1 = 0.f, s2 = 0.f;
    for (int i = tid; i < 16 * HW_; i += 256) {
        float v = xb[i];
        s1 += v; s2 += v * v;
    }
    r1[tid] = s1; r2[tid] = s2;
    __syncthreads();
    for (int off = 128; off > 0; off >>= 1) {
        if (tid < off) { r1[tid] += r1[tid + off]; r2[tid] += r2[tid + off]; }
        __syncthreads();
    }
    const float mean = r1[0] * (1.f / 16384.f);
    const float var  = r2[0] * (1.f / 16384.f) - mean * mean;
    const float rstd = rsqrtf(var + 1e-5f);

    for (int i = tid; i < 16 * HW_; i += 256) {
        const int cl = i >> 10;           // local channel 0..15
        const int p  = i & 1023;
        const int c  = g * 16 + cl;
        const float xn = (xb[i] - mean) * rstd * gamma[c] + beta[c];
        xh[((size_t)b * HW_ + p) * C_ + c] = (_Float16)xn;
    }
}

// ---------------------------------------------------------------------------
// Kernel 2: fp32 -> f16 weight conversion
// ---------------------------------------------------------------------------
__global__ void cvt_kernel(const float* __restrict__ src,
                           _Float16* __restrict__ dst, int n) {
    const int i = blockIdx.x * 256 + threadIdx.x;
    if (i < n) dst[i] = (_Float16)src[i];
}

// ---------------------------------------------------------------------------
// Shared 128x128 GEMM tile body:  C[M,N] = A[M,K] * B[N,K]^T  (both row-major)
// Double-buffered LDS + async staging.  8 waves in a 4x2 grid; each wave owns
// a 32x64 sub-tile (2 A-frags x 4 B-frags = 8 WMMAs/K-step).  B-frag loads are
// hand-interleaved so every ds_load has two independent WMMAs of shadow and
// consecutive WMMAs never share a D register (no RAW v_nop padding).
// acc index = mi*4 + n   (mi: M-subtile 0..1, n: N-subtile 0..3)
// ---------------------------------------------------------------------------
__device__ __forceinline__ void gemm_tile(const _Float16* __restrict__ A,
                                          const _Float16* __restrict__ B,
                                          int K, int m_blk, int n_blk,
                                          _Float16* Alds, _Float16* Blds,
                                          float8* acc) {
    const int wave = threadIdx.x >> 5;
    const int lane = threadIdx.x & 31;
    const int l16  = lane & 15;
    const int hv   = lane >> 4;
    const int wr   = wave >> 1;          // 0..3 : 32-row strip
    const int wc   = (wave & 1) << 6;    // 0/64 : 64-col strip
    const int nk   = K >> 5;

    // prologue: stage tile 0 into buffer 0
    async_tile_load(A, K, m_blk, 0, Alds);
    async_tile_load(B, K, n_blk, 0, Blds);
    async_wait0();
    __syncthreads();

    for (int i = 0; i < nk; ++i) {
        const int cur = (i & 1) * TILE_HALFS;
        if (i + 1 < nk) {     // kick off next tile while computing this one
            const int nxt = ((i + 1) & 1) * TILE_HALFS;
            async_tile_load(A, K, m_blk, (i + 1) * 32, Alds + nxt);
            async_tile_load(B, K, n_blk, (i + 1) * 32, Blds + nxt);
        }
        const _Float16* Ab = Alds + cur + (wr * 32) * LDS_STRIDE;
        const _Float16* Bb = Blds + cur + wc * LDS_STRIDE;

        const half16 af0 = load_frag_row(Ab,                   LDS_STRIDE, l16, hv);
        const half16 af1 = load_frag_row(Ab + 16 * LDS_STRIDE, LDS_STRIDE, l16, hv);
        half16 b0 = load_frag_row(Bb,                   LDS_STRIDE, l16, hv);
        half16 b1 = load_frag_row(Bb + 16 * LDS_STRIDE, LDS_STRIDE, l16, hv);

        acc[0] = wmma_f16(af0, b0, acc[0]);
        acc[4] = wmma_f16(af1, b0, acc[4]);
        half16 b2 = load_frag_row(Bb + 32 * LDS_STRIDE, LDS_STRIDE, l16, hv);
        acc[1] = wmma_f16(af0, b1, acc[1]);
        acc[5] = wmma_f16(af1, b1, acc[5]);
        half16 b3 = load_frag_row(Bb + 48 * LDS_STRIDE, LDS_STRIDE, l16, hv);
        acc[2] = wmma_f16(af0, b2, acc[2]);
        acc[6] = wmma_f16(af1, b2, acc[6]);
        acc[3] = wmma_f16(af0, b3, acc[3]);
        acc[7] = wmma_f16(af1, b3, acc[7]);

        async_wait0();        // next tile fully in LDS
        __syncthreads();      // everyone done reading cur / writing nxt
    }
}

// ---------------------------------------------------------------------------
// Kernel 3: QKV projection. A = xh[b] [1024,512], B = w_qkv [1536,512].
// Scatter into Q[b,h,p,d], K[b,h,p,d], Vt[b,h,d,p] (f16).
// ---------------------------------------------------------------------------
__global__ void qkv_kernel(const _Float16* __restrict__ Xh,
                           const _Float16* __restrict__ W,
                           _Float16* __restrict__ Q,
                           _Float16* __restrict__ Kb,
                           _Float16* __restrict__ Vt) {
    __shared__ alignas(16) _Float16 Alds[2 * TILE_HALFS];
    __shared__ alignas(16) _Float16 Blds[2 * TILE_HALFS];
    const int b     = blockIdx.z;
    const int m_blk = blockIdx.y * 128;
    const int n_blk = blockIdx.x * 128;
    const _Float16* A = Xh + (size_t)b * HW_ * C_;

    float8 acc[8];
#pragma unroll
    for (int n = 0; n < 8; ++n) { float8 z = {}; acc[n] = z; }

    gemm_tile(A, W, C_, m_blk, n_blk, Alds, Blds, acc);

    const int lane = threadIdx.x & 31;
    const int wave = threadIdx.x >> 5;
    const int l16  = lane & 15;
    const int hv   = lane >> 4;
    const int wr   = wave >> 1;
    const int wc   = (wave & 1) << 6;
#pragma unroll
    for (int mi = 0; mi < 2; ++mi) {
#pragma unroll
        for (int n = 0; n < 4; ++n) {
            const int o   = n_blk + wc + n * 16 + l16;
            const int seg = o >> 9;          // 0:q 1:k 2:v
            const int oo  = o & 511;
            const int hh  = oo >> 6;
            const int dd  = oo & 63;
            const size_t bh = (size_t)(b * HEADS_ + hh);
#pragma unroll
            for (int r = 0; r < 8; ++r) {
                const int p = m_blk + wr * 32 + mi * 16 + r + 8 * hv;
                const _Float16 v = (_Float16)acc[mi * 4 + n][r];
                if (seg == 0)      Q [(bh * HW_ + p) * DH_ + dd] = v;
                else if (seg == 1) Kb[(bh * HW_ + p) * DH_ + dd] = v;
                else               Vt[(bh * DH_ + dd) * HW_ + p] = v; // transposed
            }
        }
    }
}

// ---------------------------------------------------------------------------
// Kernel 4: flash attention. One block per (128 query rows, b*h); each wave
// owns 16 rows, streams keys in 32-wide tiles with online softmax.
// ---------------------------------------------------------------------------
__global__ void attn_kernel(const _Float16* __restrict__ Q,
                            const _Float16* __restrict__ Kb,
                            const _Float16* __restrict__ Vt,
                            _Float16* __restrict__ AO) {
    __shared__ alignas(16) _Float16 Plds[8 * 16 * 32];  // per-wave P tiles
    const int bh   = blockIdx.y;
    const int b    = bh >> 3;
    const int h    = bh & 7;
    const int wave = threadIdx.x >> 5;
    const int lane = threadIdx.x & 31;
    const int l16  = lane & 15;
    const int hv   = lane >> 4;
    const int q0   = blockIdx.x * 128 + wave * 16;

    const _Float16* Qp = Q  + (size_t)bh * HW_ * DH_;
    const _Float16* Kp = Kb + (size_t)bh * HW_ * DH_;
    const _Float16* Vp = Vt + (size_t)bh * DH_ * HW_;
    _Float16* Pw = Plds + wave * 16 * 32;

    // resident Q fragments (d split into two K=32 slabs)
    const half16 aq0 = load_frag_row(Qp + (size_t)q0 * DH_,      DH_, l16, hv);
    const half16 aq1 = load_frag_row(Qp + (size_t)q0 * DH_ + 32, DH_, l16, hv);

    float m[8], l[8];
    float8 O[4];
#pragma unroll
    for (int r = 0; r < 8; ++r) { m[r] = -1e30f; l[r] = 0.f; }
#pragma unroll
    for (int nt = 0; nt < 4; ++nt) { float8 z = {}; O[nt] = z; }

    const float scale = 0.125f;  // 1/sqrt(64)

    for (int j0 = 0; j0 < HW_; j0 += 32) {
        // --- hoisted K fragments; S WMMAs alternate accumulators ----------
        const _Float16* Kt0 = Kp + (size_t)j0 * DH_;
        const _Float16* Kt1 = Kp + (size_t)(j0 + 16) * DH_;
        const half16 kf0 = load_frag_row(Kt0,      DH_, l16, hv);
        const half16 kf2 = load_frag_row(Kt1,      DH_, l16, hv);
        const half16 kf1 = load_frag_row(Kt0 + 32, DH_, l16, hv);
        const half16 kf3 = load_frag_row(Kt1 + 32, DH_, l16, hv);
        float8 s0 = {}, s1 = {};
        s0 = wmma_f16(aq0, kf0, s0);
        s1 = wmma_f16(aq0, kf2, s1);
        s0 = wmma_f16(aq1, kf1, s0);
        s1 = wmma_f16(aq1, kf3, s1);

        // --- issue V fragment loads early; latency hides under softmax ---
        half16 bv[4];
#pragma unroll
        for (int nt = 0; nt < 4; ++nt)
            bv[nt] = load_frag_row(Vp + (size_t)(nt * 16) * HW_ + j0,
                                   HW_, l16, hv);

        // --- online softmax per row (row r lives in VGPR r, half hv) -----
#pragma unroll
        for (int r = 0; r < 8; ++r) {
            const float sa = s0[r] * scale;
            const float sb = s1[r] * scale;
            const float rm = redmax16(fmaxf(sa, sb));
            const float mn = fmaxf(m[r], rm);
            const float alpha = __expf(m[r] - mn);
            const float pa = __expf(sa - mn);
            const float pb = __expf(sb - mn);
            const float rs = redsum16(pa + pb);
            l[r] = l[r] * alpha + rs;
            m[r] = mn;
#pragma unroll
            for (int nt = 0; nt < 4; ++nt) O[nt][r] *= alpha;
            const int prow = r + 8 * hv;
            Pw[prow * 32 + l16]      = (_Float16)pa;  // C-layout -> LDS
            Pw[prow * 32 + 16 + l16] = (_Float16)pb;
        }
        // --- reload P as A-operand (16x32), 4 independent PV WMMAs -------
        const half16 ap = load_frag_row(Pw, 32, l16, hv);
        O[0] = wmma_f16(ap, bv[0], O[0]);
        O[1] = wmma_f16(ap, bv[1], O[1]);
        O[2] = wmma_f16(ap, bv[2], O[2]);
        O[3] = wmma_f16(ap, bv[3], O[3]);
    }

    // normalize and write [b, p, h*64+d] f16 (token-major for out-proj GEMM)
#pragma unroll
    for (int r = 0; r < 8; ++r) {
        const int p = q0 + r + 8 * hv;
        const float inv = 1.f / l[r];
#pragma unroll
        for (int nt = 0; nt < 4; ++nt) {
            AO[((size_t)b * HW_ + p) * INNER_ + h * DH_ + nt * 16 + l16] =
                (_Float16)(O[nt][r] * inv);
        }
    }
}

// ---------------------------------------------------------------------------
// Kernel 5: output projection + residual, fp32 out in [b, c, h, w]
// ---------------------------------------------------------------------------
__global__ void out_kernel(const _Float16* __restrict__ AO,
                           const _Float16* __restrict__ W,
                           const float* __restrict__ xres,
                           float* __restrict__ outp) {
    __shared__ alignas(16) _Float16 Alds[2 * TILE_HALFS];
    __shared__ alignas(16) _Float16 Blds[2 * TILE_HALFS];
    const int b     = blockIdx.z;
    const int m_blk = blockIdx.y * 128;
    const int n_blk = blockIdx.x * 128;
    const _Float16* A = AO + (size_t)b * HW_ * INNER_;

    float8 acc[8];
#pragma unroll
    for (int n = 0; n < 8; ++n) { float8 z = {}; acc[n] = z; }

    gemm_tile(A, W, INNER_, m_blk, n_blk, Alds, Blds, acc);

    const int lane = threadIdx.x & 31;
    const int wave = threadIdx.x >> 5;
    const int l16  = lane & 15;
    const int hv   = lane >> 4;
    const int wr   = wave >> 1;
    const int wc   = (wave & 1) << 6;
#pragma unroll
    for (int mi = 0; mi < 2; ++mi) {
#pragma unroll
        for (int n = 0; n < 4; ++n) {
            const int o = n_blk + wc + n * 16 + l16;
#pragma unroll
            for (int r = 0; r < 8; ++r) {
                const int p = m_blk + wr * 32 + mi * 16 + r + 8 * hv;
                const size_t idx = ((size_t)b * C_ + o) * HW_ + p;
                outp[idx] = acc[mi * 4 + n][r] + xres[idx];
            }
        }
    }
}

// ---------------------------------------------------------------------------
// Host launcher
// ---------------------------------------------------------------------------
extern "C" void kernel_launch(void* const* d_in, const int* in_sizes, int n_in,
                              void* d_out, int out_size, void* d_ws, size_t ws_size,
                              hipStream_t stream) {
    const float* x     = (const float*)d_in[0];
    const float* gamma = (const float*)d_in[1];
    const float* beta  = (const float*)d_in[2];
    const float* wqkv  = (const float*)d_in[3];
    const float* wout  = (const float*)d_in[4];
    float* outp = (float*)d_out;

    // workspace layout (f16 buffers), total ~42 MB
    char* ws = (char*)d_ws;
    _Float16* xh    = (_Float16*)(ws);                       // [b,p,c]   8 MB
    _Float16* wqkvh = (_Float16*)(ws + 8388608);             // [1536,512] 1.5 MB
    _Float16* wouth = (_Float16*)(ws + 9961472);             // [512,512] 0.5 MB
    _Float16* Qb    = (_Float16*)(ws + 10485760);            // [b,h,p,d] 8 MB
    _Float16* Kb    = (_Float16*)(ws + 18874368);            // [b,h,p,d] 8 MB
    _Float16* Vt    = (_Float16*)(ws + 27262976);            // [b,h,d,p] 8 MB
    _Float16* AO    = (_Float16*)(ws + 35651584);            // [b,p,c]   8 MB

    // 1. GroupNorm + f16 convert (one block per batch*group)
    gn_kernel<<<dim3(B_ * GROUPS_), dim3(256), 0, stream>>>(x, gamma, beta, xh);

    // 2. weight conversion
    cvt_kernel<<<dim3((3 * INNER_ * C_ + 255) / 256), dim3(256), 0, stream>>>(
        wqkv, wqkvh, 3 * INNER_ * C_);
    cvt_kernel<<<dim3((C_ * INNER_ + 255) / 256), dim3(256), 0, stream>>>(
        wout, wouth, C_ * INNER_);

    // 3. QKV GEMM: M=1024, N=1536, K=512 per batch
    qkv_kernel<<<dim3(1536 / 128, HW_ / 128, B_), dim3(256), 0, stream>>>(
        xh, wqkvh, Qb, Kb, Vt);

    // 4. flash attention: grid = (query tiles of 128, b*heads)
    attn_kernel<<<dim3(HW_ / 128, B_ * HEADS_), dim3(256), 0, stream>>>(
        Qb, Kb, Vt, AO);

    // 5. out projection + residual: M=1024, N=512, K=512 per batch
    out_kernel<<<dim3(C_ / 128, HW_ / 128, B_), dim3(256), 0, stream>>>(
        AO, wouth, x, outp);
}
